// SGLMamba_15788299780393
// MI455X (gfx1250) — compile-verified, matrix operand order checked by reading, not run on hardware
//
#include <hip/hip_runtime.h>
#include <hip/hip_bf16.h>
#include <math.h>
#include <stdint.h>

typedef _Float16 h16;
typedef __attribute__((ext_vector_type(8)))  _Float16 v8h;
typedef __attribute__((ext_vector_type(16))) _Float16 v16h;
typedef __attribute__((ext_vector_type(8)))  float    v8f;
typedef __attribute__((ext_vector_type(4)))  int      v4i;

// ---------------------------------------------------------------------------
// gfx1250 async global->LDS copy (ASYNCcnt path), with safe fallback.
// Builtin signature (from clang diagnostic): (v4i AS1*, v4i AS3*, imm, imm).
// AS3 pointers are 32-bit LDS offsets == low 32 bits of the flat LDS address,
// so the integer-cast idiom is semantically correct, not just compile-legal.
// ---------------------------------------------------------------------------
#if __has_builtin(__builtin_amdgcn_global_load_async_to_lds_b128) && \
    __has_builtin(__builtin_amdgcn_s_wait_asynccnt)
#define USE_ASYNC_LDS 1
#else
#define USE_ASYNC_LDS 0
#endif

__device__ __forceinline__ void async_copy_b128(const h16* g, h16* s) {
#if USE_ASYNC_LDS
  __builtin_amdgcn_global_load_async_to_lds_b128(
      (__attribute__((address_space(1))) v4i*)(uintptr_t)g,
      (__attribute__((address_space(3))) v4i*)(unsigned)(uintptr_t)s, 0, 0);
#else
  *(v8h*)s = *(const v8h*)g;
#endif
}
__device__ __forceinline__ void async_wait0() {
#if USE_ASYNC_LDS
  __builtin_amdgcn_s_wait_asynccnt(0);
#endif
}

// ---------------------------------------------------------------------------
// WMMA fragment loads (CDNA5 16x16x32 f16, wave32).
// Per lane the 16 halves are two contiguous 8-half runs:
//   k = 8*hi + [0..8)  and  k = 16 + 8*hi + [0..8)
// so each fragment is two 16-byte LDS vector loads.  A tiles are m-major
// (stride multiple of 8 halves), B tiles are staged n-major for the same
// contiguity.
// ---------------------------------------------------------------------------
__device__ __forceinline__ v16h frag_load(const h16* p) {
  v8h lo = *(const v8h*)p;
  v8h hi = *(const v8h*)(p + 16);
  return __builtin_shufflevector(lo, hi, 0, 1, 2, 3, 4, 5, 6, 7, 8, 9, 10, 11,
                                 12, 13, 14, 15);
}

// ---------------------------------------------------------------------------
// wave32 shuffle reductions; 64-wide segment = 2 waves combined through LDS.
// Dual (sum, sumsq) reduction: one LDS exchange for a full LN.
// ---------------------------------------------------------------------------
__device__ __forceinline__ void seg_reduce64_2(float a, float b, float* red,
                                               float& ra, float& rb) {
  int t = threadIdx.x, wv = t >> 5;
#pragma unroll
  for (int off = 16; off >= 1; off >>= 1) {
    a += __shfl_xor(a, off, 32);
    b += __shfl_xor(b, off, 32);
  }
  if ((t & 31) == 0) {
    red[wv * 2] = a;
    red[wv * 2 + 1] = b;
  }
  __syncthreads();
  int s = (wv & ~1) * 2;
  ra = red[s] + red[s + 2];
  rb = red[s + 1] + red[s + 3];
  __syncthreads();
}

// ---------------------------------------------------------------------------
// K1: per-pixel LN over C (64), 2*x-1, Haar DWT.
// Block = 256 threads = 4 output pixels x 64 channels.
// ---------------------------------------------------------------------------
__global__ __launch_bounds__(256) void k_ln1_dwt(
    const float* __restrict__ x, const float* __restrict__ w1,
    const float* __restrict__ b1, float* __restrict__ LL,
    float* __restrict__ hr) {
  __shared__ float red[16];
  size_t p = (size_t)blockIdx.x * 4 + (threadIdx.x >> 6);
  int c = threadIdx.x & 63;
  int w2 = (int)(p & 127);
  int h2 = (int)((p >> 7) & 127);
  int b = (int)(p >> 14);
  float q[4];
#pragma unroll
  for (int j = 0; j < 4; ++j) {
    int dy = j >> 1, dx = j & 1;
    int y = 2 * h2 + dy, xx = 2 * w2 + dx;
    float v = x[(((size_t)(b * 64 + c)) << 16) + (size_t)y * 256 + xx];
    float sum, ssq;
    seg_reduce64_2(v, v * v, red, sum, ssq);
    float mu = sum * (1.f / 64.f);
    float var = fmaxf(ssq * (1.f / 64.f) - mu * mu, 0.f);
    float lnv = (v - mu) * rsqrtf(var + 1e-5f) * w1[c] + b1[c];
    q[j] = (2.f * lnv - 1.f) * 0.5f;  // includes the /2 inside dwt
  }
  float x1 = q[0], x3 = q[1], x2 = q[2], x4 = q[3];
  size_t l = (size_t)h2 * 128 + w2;
  LL[(((size_t)b * 16384 + l) << 6) + c] = x1 + x2 + x3 + x4;
  hr[((((size_t)(0 + b)) * 16384 + l) << 6) + c] = -x1 - x2 + x3 + x4;
  hr[((((size_t)(2 + b)) * 16384 + l) << 6) + c] = -x1 + x2 - x3 + x4;
  hr[((((size_t)(4 + b)) * 16384 + l) << 6) + c] = x1 - x2 - x3 + x4;
}

// ---------------------------------------------------------------------------
// Generic per-row (64-wide) LN -> f16 output (feeds WMMA A matrices).
// ---------------------------------------------------------------------------
__global__ __launch_bounds__(256) void k_rowln_f16(
    const float* __restrict__ in, const float* __restrict__ w,
    const float* __restrict__ b, h16* __restrict__ out, size_t nrows) {
  __shared__ float red[16];
  size_t row = (size_t)blockIdx.x * 4 + (threadIdx.x >> 6);
  int c = threadIdx.x & 63;
  float v = (row < nrows) ? in[row * 64 + c] : 0.f;
  float sum, ssq;
  seg_reduce64_2(v, v * v, red, sum, ssq);
  float mu = sum * (1.f / 64.f);
  float var = fmaxf(ssq * (1.f / 64.f) - mu * mu, 0.f);
  if (row < nrows)
    out[row * 64 + c] = (h16)((v - mu) * rsqrtf(var + 1e-5f) * w[c] + b[c]);
}

// ---------------------------------------------------------------------------
// WMMA GEMM: out[M,N] = A(f16, MxK row-major) @ Wt(f32, NxK row-major)^T.
// BM=128 / BN=64 per block (grid.y tiles N), BK=32, 8 waves; wave w owns the
// 16-row strip x 64 cols (4 accumulators).  A tiles staged with async
// global->LDS b128; weights converted f32->f16 on VALU while asyncs fly.
// EPI 0: +bias, store f32 (row stride Nld)
// EPI 1: +res[m*Nld+n], store f32
// EPI 2: +bias; fused final output: e_out NCHW = e_xres + v*e_scale + e_xn2
// ---------------------------------------------------------------------------
template <int EPI>
__global__ __launch_bounds__(256) void k_gemm(
    const h16* __restrict__ A, const float* __restrict__ Wt,
    const float* __restrict__ bias, const float* __restrict__ res,
    float* __restrict__ out, int K, int Nld,
    const float* __restrict__ e_xres, const float* __restrict__ e_xn2,
    const float* __restrict__ e_scale, float* __restrict__ e_out) {
  __shared__ __align__(16) h16 sA[128 * 40];   // m-major, stride 40 halves
  __shared__ __align__(16) h16 sBt[64 * 40];   // n-major, stride 40 halves
  const int m0 = blockIdx.x * 128;
  const int n0 = blockIdx.y * 64;
  const int t = threadIdx.x;
  const int w = t >> 5;
  const int lane = t & 31, hi = lane >> 4;
  v8f acc[4] = {};
  for (int kk = 0; kk < K; kk += 32) {
    {  // A tile: 128 rows x 32 halves = 512 b128 segments, 2 per thread
      int r = t >> 1;
      int s0 = (t & 1) * 2;
      const h16* g = A + (size_t)(m0 + r) * K + kk;
      h16* s = sA + r * 40;
      async_copy_b128(g + s0 * 8, s + s0 * 8);
      async_copy_b128(g + s0 * 8 + 8, s + s0 * 8 + 8);
      if (kk + 32 < K) __builtin_prefetch(g + 32, 0, 0);
    }
    {  // B tile (n-major): convert f32 weights -> f16
      int n = t & 63, kbase = (t >> 6) * 8;
      const float* g = Wt + (size_t)(n0 + n) * K + kk + kbase;
      h16* s = sBt + n * 40 + kbase;
#pragma unroll
      for (int i = 0; i < 8; ++i) s[i] = (h16)g[i];
    }
    async_wait0();
    __syncthreads();
    v16h a = frag_load(sA + (w * 16 + (lane & 15)) * 40 + 8 * hi);
#pragma unroll
    for (int ns = 0; ns < 4; ++ns) {
      v16h b = frag_load(sBt + (ns * 16 + (lane & 15)) * 40 + 8 * hi);
      acc[ns] = __builtin_amdgcn_wmma_f32_16x16x32_f16(
          false, a, false, b, (short)0, acc[ns], false, false);
    }
    __syncthreads();
  }
#pragma unroll
  for (int ns = 0; ns < 4; ++ns) {
    int n = n0 + ns * 16 + (lane & 15);
#pragma unroll
    for (int r = 0; r < 8; ++r) {
      int m = m0 + w * 16 + hi * 8 + r;
      float v = acc[ns][r];
      if (bias) v += bias[n];
      if (EPI == 1) v += res[(size_t)m * Nld + n];
      if (EPI == 2) {
        size_t idx = (size_t)m * 64 + n;
        float gs = v * e_scale[n] + e_xn2[idx];
        int bb = m >> 16;
        int pix = m & 0xFFFF;
        e_out[((size_t)(bb * 64 + n) << 16) + pix] = e_xres[idx] + gs;
      } else {
        out[(size_t)m * Nld + n] = v;
      }
    }
  }
}

// ---------------------------------------------------------------------------
// Depthwise causal conv1d (k=4) over L + SiLU.  xz NHWC128 -> xs NHWC64.
// ---------------------------------------------------------------------------
__global__ void k_conv1d_silu(const float* __restrict__ xz,
                              const float* __restrict__ cw,
                              const float* __restrict__ cb,
                              float* __restrict__ xs) {
  size_t i = (size_t)blockIdx.x * 256 + threadIdx.x;  // 6*16384*64
  if (i >= (size_t)6 * 16384 * 64) return;
  int c = (int)(i & 63);
  size_t tok = i >> 6;  // b*16384 + l
  int l = (int)(tok & 16383);
  float acc = cb[c];
#pragma unroll
  for (int k = 0; k < 4; ++k) {
    int ll = l - 3 + k;
    if (ll >= 0) acc += xz[(tok + (size_t)(ll - l)) * 128 + c] * cw[c * 4 + k];
  }
  xs[i] = acc / (1.f + expf(-acc));
}

// ---------------------------------------------------------------------------
// Per-token x-proj (20 outs) + dt (softplus), split B/C.
// ---------------------------------------------------------------------------
__global__ void k_xproj_dt(const float* __restrict__ xs,
                           const float* __restrict__ xpw,
                           const float* __restrict__ dtw,
                           const float* __restrict__ dtb,
                           float* __restrict__ dt, float* __restrict__ Bm,
                           float* __restrict__ Cm) {
  size_t tok = (size_t)blockIdx.x * 256 + threadIdx.x;
  if (tok >= (size_t)98304) return;
  const float* row = xs + tok * 64;
  float xr[64];
#pragma unroll
  for (int c = 0; c < 64; ++c) xr[c] = row[c];
  float xdb[20];
#pragma unroll 4
  for (int j = 0; j < 20; ++j) {
    float a = 0.f;
#pragma unroll
    for (int c = 0; c < 64; ++c) a += xr[c] * xpw[j * 64 + c];
    xdb[j] = a;
  }
#pragma unroll
  for (int n = 0; n < 8; ++n) {
    Bm[tok * 8 + n] = xdb[4 + n];
    Cm[tok * 8 + n] = xdb[12 + n];
  }
#pragma unroll 4
  for (int d = 0; d < 64; ++d) {
    float a = dtb[d];
#pragma unroll
    for (int r = 0; r < 4; ++r) a += xdb[r] * dtw[d * 4 + r];
    dt[tok * 64 + d] = (a > 20.f) ? a : log1pf(expf(a));
  }
}

// ---------------------------------------------------------------------------
// Selective scan. One block per band-batch (6), 64 threads = channels d.
// 8-state recurrence in registers; B/C rows fetched as float4 broadcasts;
// fused +xs*D, output LN (oln), * silu(z).  One barrier per timestep via a
// parity double-buffered 8-float LDS slab.
// ---------------------------------------------------------------------------
__global__ __launch_bounds__(64) void k_scan(
    const float* __restrict__ dt, const float* __restrict__ xs,
    const float* __restrict__ Bm, const float* __restrict__ Cm,
    const float* __restrict__ xz, const float* __restrict__ Alog,
    const float* __restrict__ Dp, const float* __restrict__ olnw,
    const float* __restrict__ olnb, h16* __restrict__ yg) {
  __shared__ float red[8];
  int b = blockIdx.x, d = threadIdx.x;
  int wv = d >> 5, lane = d & 31;
  float A[8], h[8];
#pragma unroll
  for (int n = 0; n < 8; ++n) {
    A[n] = -expf(Alog[d * 8 + n]);
    h[n] = 0.f;
  }
  float Dd = Dp[d], wln = olnw[d], bln = olnb[d];
  for (int t = 0; t < 16384; ++t) {
    size_t tok = (size_t)b * 16384 + t;
    const float4* bp = (const float4*)(Bm + tok * 8);
    const float4* cp = (const float4*)(Cm + tok * 8);
    float4 b0 = bp[0], b1 = bp[1];
    float4 c0 = cp[0], c1 = cp[1];
    float sBv[8] = {b0.x, b0.y, b0.z, b0.w, b1.x, b1.y, b1.z, b1.w};
    float sCv[8] = {c0.x, c0.y, c0.z, c0.w, c1.x, c1.y, c1.z, c1.w};
    float dtv = dt[tok * 64 + d];
    float xv = xs[tok * 64 + d];
    float dBx = dtv * xv;
    float y = 0.f;
#pragma unroll
    for (int n = 0; n < 8; ++n) {
      h[n] = expf(dtv * A[n]) * h[n] + dBx * sBv[n];
      y += h[n] * sCv[n];
    }
    y += xv * Dd;
    float s1 = y, s2 = y * y;
#pragma unroll
    for (int off = 16; off >= 1; off >>= 1) {
      s1 += __shfl_xor(s1, off, 32);
      s2 += __shfl_xor(s2, off, 32);
    }
    int bank = (t & 1) * 4;
    if (lane == 0) {
      red[bank + wv * 2] = s1;
      red[bank + wv * 2 + 1] = s2;
    }
    __syncthreads();
    float sum = red[bank + 0] + red[bank + 2];
    float ssq = red[bank + 1] + red[bank + 3];
    float mu = sum * (1.f / 64.f);
    float var = fmaxf(ssq * (1.f / 64.f) - mu * mu, 0.f);
    float yn = (y - mu) * rsqrtf(var + 1e-5f) * wln + bln;
    float zv = xz[tok * 128 + 64 + d];
    float g = zv / (1.f + expf(-zv));
    yg[tok * 64 + d] = (h16)(yn * g);
  }
}

// ---------------------------------------------------------------------------
// Implicit-GEMM 3x3 conv (C=64 -> 64, pad 1) + leaky ReLU, NHWC.
// Halo 3x34x64 f16 in LDS; 18 k-chunks (9 taps x 64ch) through WMMA.
// ---------------------------------------------------------------------------
__global__ __launch_bounds__(256) void k_conv3x3(
    const float* __restrict__ src, const float* __restrict__ wgt,
    const float* __restrict__ bias, float* __restrict__ dst) {
  __shared__ __align__(16) h16 halo[3 * 34 * 64];
  __shared__ __align__(16) h16 sBt[64 * 40];  // n-major
  const int w0 = blockIdx.x * 32;
  const int h = blockIdx.y;
  const int bb = blockIdx.z;
  const int t = threadIdx.x;
  for (int idx = t; idx < 3 * 34 * 64; idx += 256) {
    int c = idx & 63;
    int colrow = idx >> 6;
    int col = colrow % 34;
    int row = colrow / 34;
    int y = h + row - 1, xx = w0 + col - 1;
    float v = 0.f;
    if (y >= 0 && y < 128 && xx >= 0 && xx < 128)
      v = src[(((size_t)(bb * 128 + y)) * 128 + xx) * 64 + c];
    halo[idx] = (h16)v;
  }
  const int wv = t >> 5;
  const int msub = wv & 1, nsub = wv >> 1;
  int lane = t & 31, hi = lane >> 4;
  v8f acc = {};
  for (int kkq = 0; kkq < 18; ++kkq) {
    int tap = kkq >> 1, ci0 = (kkq & 1) * 32;
    int dy = tap / 3, dx = tap % 3;
    {  // stage weights for this tap/channel-half, n-major
      int n = t & 63, kbase = (t >> 6) * 8;
      h16* s = sBt + n * 40 + kbase;
#pragma unroll
      for (int i = 0; i < 8; ++i)
        s[i] = (h16)wgt[((size_t)n * 64 + (ci0 + kbase + i)) * 9 + tap];
    }
    __syncthreads();
    v16h a = frag_load(halo + ((dy * 34) + (msub * 16 + (lane & 15) + dx)) * 64 +
                       ci0 + 8 * hi);
    v16h bmat = frag_load(sBt + (nsub * 16 + (lane & 15)) * 40 + 8 * hi);
    acc = __builtin_amdgcn_wmma_f32_16x16x32_f16(false, a, false, bmat,
                                                 (short)0, acc, false, false);
    __syncthreads();
  }
#pragma unroll
  for (int r = 0; r < 8; ++r) {
    int n = nsub * 16 + (lane & 15);
    int m = msub * 16 + hi * 8 + r;
    float v = acc[r] + bias[n];
    v = (v >= 0.f) ? v : 0.1f * v;
    dst[(((size_t)(bb * 128 + h)) * 128 + (w0 + m)) * 64 + n] = v;
  }
}

// ---------------------------------------------------------------------------
// IWT + clip + residual add (original x, NCHW). Writes xres NHWC (2,256,256,64).
// ---------------------------------------------------------------------------
__global__ void k_iwt(const float* __restrict__ y3, const float* __restrict__ hm,
                      const float* __restrict__ xin, float* __restrict__ xres) {
  size_t i = (size_t)blockIdx.x * 256 + threadIdx.x;  // 2*128*128*64
  if (i >= (size_t)2 * 128 * 128 * 64) return;
  int c = (int)(i & 63);
  size_t p = i >> 6;
  int w2 = (int)(p & 127);
  int h2 = (int)((p >> 7) & 127);
  int b = (int)(p >> 14);
  float a1 = 0.5f * y3[i];  // y3 is NHWC, same flat index
  size_t hb = (size_t)c * 16384 + (size_t)h2 * 128 + w2;  // hm as NCHW
  float a2 = 0.5f * hm[(((size_t)(0 + b)) << 20) + hb];
  float a3 = 0.5f * hm[(((size_t)(2 + b)) << 20) + hb];
  float a4 = 0.5f * hm[(((size_t)(4 + b)) << 20) + hb];
  float v00 = a1 - a2 - a3 + a4;
  float v10 = a1 - a2 + a3 - a4;
  float v01 = a1 + a2 - a3 - a4;
  float v11 = a1 + a2 + a3 + a4;
  int y0 = 2 * h2, x0 = 2 * w2;
  auto emit = [&](int y, int xx, float v) {
    float xw = fminf(fmaxf((v + 1.f) * 0.5f, 0.f), 1.f);
    float sc = xin[(((size_t)(b * 64 + c)) << 16) + (size_t)y * 256 + xx];
    xres[((((size_t)(b * 256 + y)) * 256 + xx) << 6) + c] = xw + sc;
  };
  emit(y0, x0, v00);
  emit(y0 + 1, x0, v10);
  emit(y0, x0 + 1, v01);
  emit(y0 + 1, x0 + 1, v11);
}

// ---------------------------------------------------------------------------
// Fused LN2 (eps 1e-5) -> xn2 f32, then per-pixel channel norm with g_ln
// (eps 1e-6) -> gn f16 (A matrix for g_c1).
// ---------------------------------------------------------------------------
__global__ __launch_bounds__(256) void k_norm2(
    const float* __restrict__ xres, const float* __restrict__ w2,
    const float* __restrict__ b2, const float* __restrict__ gw,
    const float* __restrict__ gb, float* __restrict__ xn2,
    h16* __restrict__ gn) {
  __shared__ float red[16];
  size_t row = (size_t)blockIdx.x * 4 + (threadIdx.x >> 6);
  int c = threadIdx.x & 63;
  float v = xres[row * 64 + c];
  float sum, ssq;
  seg_reduce64_2(v, v * v, red, sum, ssq);
  float mu = sum * (1.f / 64.f);
  float var = fmaxf(ssq * (1.f / 64.f) - mu * mu, 0.f);
  float x2v = (v - mu) * rsqrtf(var + 1e-5f) * w2[c] + b2[c];
  xn2[row * 64 + c] = x2v;
  float sum2, ssq2;
  seg_reduce64_2(x2v, x2v * x2v, red, sum2, ssq2);
  float mu2 = sum2 * (1.f / 64.f);
  float var2 = fmaxf(ssq2 * (1.f / 64.f) - mu2 * mu2, 0.f);
  gn[row * 64 + c] = (h16)((x2v - mu2) * rsqrtf(var2 + 1e-6f) * gw[c] + gb[c]);
}

// ---------------------------------------------------------------------------
// Depthwise 7x7 on a = t[:, :64], multiply by v = t[:, 64:], f16 out (g_c2 A).
// ---------------------------------------------------------------------------
__global__ void k_dw7x7(const float* __restrict__ tbuf,
                        const float* __restrict__ dw,
                        const float* __restrict__ db, h16* __restrict__ vp) {
  size_t i = (size_t)blockIdx.x * 256 + threadIdx.x;  // 2*256*256*64
  if (i >= (size_t)2 * 256 * 256 * 64) return;
  int c = (int)(i & 63);
  size_t p = i >> 6;
  int x = (int)(p & 255);
  int y = (int)((p >> 8) & 255);
  int b = (int)(p >> 16);
  float acc = db[c];
#pragma unroll
  for (int dy = -3; dy <= 3; ++dy) {
    int yy = y + dy;
    if (yy < 0 || yy > 255) continue;
#pragma unroll
    for (int dx = -3; dx <= 3; ++dx) {
      int xx = x + dx;
      if (xx < 0 || xx > 255) continue;
      float av = tbuf[((((size_t)(b * 256 + yy)) * 256 + xx) << 7) + c];
      acc += av * dw[c * 49 + (dy + 3) * 7 + (dx + 3)];
    }
  }
  float vv = tbuf[((((size_t)(b * 256 + y)) * 256 + x) << 7) + 64 + c];
  vp[i] = (h16)(vv * acc);
}

// ---------------------------------------------------------------------------
// Host side
// ---------------------------------------------------------------------------
extern "C" void kernel_launch(void* const* d_in, const int* in_sizes, int n_in,
                              void* d_out, int out_size, void* d_ws,
                              size_t ws_size, hipStream_t stream) {
  const float* x        = (const float*)d_in[0];
  const float* ln1_w    = (const float*)d_in[1];
  const float* ln1_b    = (const float*)d_in[2];
  const float* m_ln_w   = (const float*)d_in[3];
  const float* m_ln_b   = (const float*)d_in[4];
  const float* m_in_w   = (const float*)d_in[5];
  const float* m_in_b   = (const float*)d_in[6];
  const float* m_conv_w = (const float*)d_in[7];
  const float* m_conv_b = (const float*)d_in[8];
  const float* m_xproj_w= (const float*)d_in[9];
  const float* m_dt_w   = (const float*)d_in[10];
  const float* m_dt_b   = (const float*)d_in[11];
  const float* m_Alog   = (const float*)d_in[12];
  const float* m_D      = (const float*)d_in[13];
  const float* m_oln_w  = (const float*)d_in[14];
  const float* m_oln_b  = (const float*)d_in[15];
  const float* m_out_w  = (const float*)d_in[16];
  const float* p1_w1    = (const float*)d_in[17];
  const float* p1_b1    = (const float*)d_in[18];
  const float* p1_w2    = (const float*)d_in[19];
  const float* p1_b2    = (const float*)d_in[20];
  const float* p1_w3    = (const float*)d_in[21];
  const float* p1_b3    = (const float*)d_in[22];
  const float* ln2_w    = (const float*)d_in[23];
  const float* ln2_b    = (const float*)d_in[24];
  const float* g_ln_w   = (const float*)d_in[25];
  const float* g_ln_b   = (const float*)d_in[26];
  const float* g_c1_w   = (const float*)d_in[27];
  const float* g_c1_b   = (const float*)d_in[28];
  const float* g_dw_w   = (const float*)d_in[29];
  const float* g_dw_b   = (const float*)d_in[30];
  const float* g_c2_w   = (const float*)d_in[31];
  const float* g_c2_b   = (const float*)d_in[32];
  const float* g_scale  = (const float*)d_in[33];

  char* ws = (char*)d_ws;
  size_t off = 0;
  auto take = [&](size_t bytes) -> void* {
    void* p = ws + off;
    off = (off + bytes + 255) & ~(size_t)255;
    return p;
  };
  float* LL   = (float*)take((size_t)2 * 128 * 128 * 64 * 4);
  float* hr   = (float*)take((size_t)6 * 16384 * 64 * 4);
  h16*   u    = (h16*)  take((size_t)6 * 16384 * 64 * 2);
  float* xz   = (float*)take((size_t)6 * 16384 * 128 * 4);
  float* xs   = (float*)take((size_t)6 * 16384 * 64 * 4);
  float* dtb  = (float*)take((size_t)6 * 16384 * 64 * 4);
  float* Bm   = (float*)take((size_t)6 * 16384 * 8 * 4);
  float* Cm   = (float*)take((size_t)6 * 16384 * 8 * 4);
  h16*   yg   = (h16*)  take((size_t)6 * 16384 * 64 * 2);
  float* hm   = (float*)take((size_t)6 * 16384 * 64 * 4);
  float* y1   = (float*)take((size_t)2 * 128 * 128 * 64 * 4);
  float* y2   = (float*)take((size_t)2 * 128 * 128 * 64 * 4);
  float* y3   = (float*)take((size_t)2 * 128 * 128 * 64 * 4);
  float* xres = (float*)take((size_t)2 * 256 * 256 * 64 * 4);
  float* xn2  = (float*)take((size_t)2 * 256 * 256 * 64 * 4);
  h16*   gn   = (h16*)  take((size_t)2 * 256 * 256 * 64 * 2);
  float* tb   = (float*)take((size_t)2 * 256 * 256 * 128 * 4);
  h16*   vp   = (h16*)  take((size_t)2 * 256 * 256 * 64 * 2);

  // 1) LN1 + DWT
  k_ln1_dwt<<<8192, 256, 0, stream>>>(x, ln1_w, ln1_b, LL, hr);
  // 2) mamba input LN -> f16
  k_rowln_f16<<<24576, 256, 0, stream>>>(hr, m_ln_w, m_ln_b, u, (size_t)98304);
  // 3) in_proj: xz = u @ in_w^T + b   (M=98304, K=64, N=128)
  k_gemm<0><<<dim3(768, 2), 256, 0, stream>>>(u, m_in_w, m_in_b, nullptr, xz,
                                              64, 128, nullptr, nullptr,
                                              nullptr, nullptr);
  // 4) causal dwconv1d + silu
  k_conv1d_silu<<<24576, 256, 0, stream>>>(xz, m_conv_w, m_conv_b, xs);
  // 5) x-proj + dt
  k_xproj_dt<<<384, 256, 0, stream>>>(xs, m_xproj_w, m_dt_w, m_dt_b, dtb, Bm, Cm);
  // 6) selective scan + oln + gate -> yg (f16)
  k_scan<<<6, 64, 0, stream>>>(dtb, xs, Bm, Cm, xz, m_Alog, m_D, m_oln_w,
                               m_oln_b, yg);
  // 7) out_proj + residual hr -> hm (== out_high flat)
  k_gemm<1><<<dim3(768, 1), 256, 0, stream>>>(yg, m_out_w, nullptr, hr, hm, 64,
                                              64, nullptr, nullptr, nullptr,
                                              nullptr);
  // 8) three 3x3 convs + leaky relu on LL
  k_conv3x3<<<dim3(4, 128, 2), 256, 0, stream>>>(LL, p1_w1, p1_b1, y1);
  k_conv3x3<<<dim3(4, 128, 2), 256, 0, stream>>>(y1, p1_w2, p1_b2, y2);
  k_conv3x3<<<dim3(4, 128, 2), 256, 0, stream>>>(y2, p1_w3, p1_b3, y3);
  // 9) IWT + clip + residual -> xres (NHWC)
  k_iwt<<<8192, 256, 0, stream>>>(y3, hm, x, xres);
  // 10) LN2 + channel norm -> xn2 (f32), gn (f16)
  k_norm2<<<32768, 256, 0, stream>>>(xres, ln2_w, ln2_b, g_ln_w, g_ln_b, xn2, gn);
  // 11) g_c1 1x1 conv (M=131072, K=64, N=128)
  k_gemm<0><<<dim3(1024, 2), 256, 0, stream>>>(gn, g_c1_w, g_c1_b, nullptr, tb,
                                               64, 128, nullptr, nullptr,
                                               nullptr, nullptr);
  // 12) depthwise 7x7 gate
  k_dw7x7<<<32768, 256, 0, stream>>>(tb, g_dw_w, g_dw_b, vp);
  // 13) g_c2 1x1 conv + fused final output (NCHW): out = xres + v*scale + xn2
  k_gemm<2><<<dim3(1024, 1), 256, 0, stream>>>(vp, g_c2_w, g_c2_b, nullptr,
                                               nullptr, 64, 64, xres, xn2,
                                               g_scale, (float*)d_out);
}
